// SCModule_4844723110562
// MI455X (gfx1250) — matrix operand order repaired and checked by reading, not compiled, not run on hardware
//
#include <hip/hip_runtime.h>
#include <hip/hip_bf16.h>
#include <math.h>

// Problem sizes (fixed by the reference)
#define Bsz 8
#define Cch 256
#define Nn  4096
#define Dd  32
#define Rr  32

typedef __attribute__((ext_vector_type(16))) __bf16 v16bf;
typedef __attribute__((ext_vector_type(8)))  __bf16 v8bf;
typedef __attribute__((ext_vector_type(4)))  __bf16 v4bf;
typedef __attribute__((ext_vector_type(8)))  float  v8f;

static __device__ __forceinline__ v8f wmma_bf16(v16bf a, v16bf b, v8f c) {
  // D = A(16x32 bf16) x B(32x16 bf16) + C(16x16 f32)
  return __builtin_amdgcn_wmma_f32_16x16x32_bf16(false, a, false, b, (short)0, c,
                                                 false, false);
}

static __device__ __forceinline__ v16bf cat8(v8bf lo, v8bf hi) {
  return __builtin_shufflevector(lo, hi, 0, 1, 2, 3, 4, 5, 6, 7,
                                 8, 9, 10, 11, 12, 13, 14, 15);
}

// ---------------------------------------------------------------------------
// Kernel 0: one-shot f32 -> bf16 conversion (weights; reused by all blocks).
// ---------------------------------------------------------------------------
__global__ void cvt_bf16_kernel(const float* __restrict__ src,
                                __bf16* __restrict__ dst, int n) {
  int i = (blockIdx.x * 256 + threadIdx.x) * 4;
  if (i < n) {
    float4 v = *(const float4*)(src + i);
    v4bf o;
    o[0] = (__bf16)v.x; o[1] = (__bf16)v.y;
    o[2] = (__bf16)v.z; o[3] = (__bf16)v.w;
    *(v4bf*)(dst + i) = o;
  }
}

// ---------------------------------------------------------------------------
// Kernel 1: per-(b,c) mean and unbiased std over N=4096 spatial elements.
// ---------------------------------------------------------------------------
__global__ void stats_kernel(const float* __restrict__ x,
                             float* __restrict__ meanv,
                             float* __restrict__ stdv) {
  int bc = blockIdx.x;                       // 0..B*C-1
  const float4* row4 = (const float4*)(x + (size_t)bc * Nn);
  float s = 0.f, ss = 0.f;
  for (int i = threadIdx.x; i < Nn / 4; i += 256) {
    float4 v = row4[i];
    s  += v.x + v.y + v.z + v.w;
    ss += v.x * v.x + v.y * v.y + v.z * v.z + v.w * v.w;
  }
  __shared__ float sb[256], qb_[256];
  sb[threadIdx.x] = s; qb_[threadIdx.x] = ss;
  __syncthreads();
  for (int off = 128; off > 0; off >>= 1) {
    if (threadIdx.x < off) {
      sb[threadIdx.x]  += sb[threadIdx.x + off];
      qb_[threadIdx.x] += qb_[threadIdx.x + off];
    }
    __syncthreads();
  }
  if (threadIdx.x == 0) {
    float m   = sb[0] / (float)Nn;
    float var = (qb_[0] - (float)Nn * m * m) / (float)(Nn - 1);   // ddof=1
    meanv[bc] = m;
    stdv[bc]  = sqrtf(fmaxf(var, 0.f));
  }
}

// ---------------------------------------------------------------------------
// Kernel 2: SE gate MLP.  g[b,c] = 1 + lamb * sigmoid(relu([mean,std]W1^T)W2^T)
// ---------------------------------------------------------------------------
__global__ void se_kernel(const float* __restrict__ meanv,
                          const float* __restrict__ stdv,
                          const float* __restrict__ W1,
                          const float* __restrict__ W2,
                          const float* __restrict__ lamb,
                          float* __restrict__ g) {
  int b   = blockIdx.x;
  int tid = threadIdx.x;                     // 256 threads
  __shared__ float inp[2 * Cch];
  __shared__ float hid[Rr];
  inp[tid]        = meanv[b * Cch + tid];
  inp[Cch + tid]  = stdv[b * Cch + tid];
  __syncthreads();
  if (tid < Rr) {
    float acc = 0.f;
    for (int j = 0; j < 2 * Cch; ++j) acc += W1[tid * 2 * Cch + j] * inp[j];
    hid[tid] = fmaxf(acc, 0.f);
  }
  __syncthreads();
  float acc = 0.f;
  for (int r = 0; r < Rr; ++r) acc += W2[tid * Rr + r] * hid[r];
  float a = 1.f / (1.f + expf(-acc));
  g[b * Cch + tid] = 1.0f + lamb[0] * a;
}

// Double-buffered staging of a 32c x 16n f32 tile into transposed padded LDS
// [n16][c32->40] bf16.  128 threads: exactly one float4 load each.
static __device__ __forceinline__ float4 load_xquad(const float* __restrict__ xsrc,
                                                    int k0, int n0, int tid) {
  int c  = tid >> 2;
  int nq = (tid & 3) * 4;
  return *(const float4*)(xsrc + (size_t)(k0 + c) * Nn + n0 + nq);
}
static __device__ __forceinline__ void store_xquad(float4 v, int tid,
                                                   __bf16 (*xa)[40]) {
  int c  = tid >> 2;
  int nq = (tid & 3) * 4;
  xa[nq + 0][c] = (__bf16)v.x;
  xa[nq + 1][c] = (__bf16)v.y;
  xa[nq + 2][c] = (__bf16)v.z;
  xa[nq + 3][c] = (__bf16)v.w;
}

// ---------------------------------------------------------------------------
// Kernel 3: q & k projections.  q[b,n,d] = sum_c Wq[d,c]*x[b,c,n] + bq[d]
// Weights pre-converted to bf16.  Block = 128 threads (4 waves):
//   wave 0/1 -> q d-tiles 0/1, wave 2/3 -> k d-tiles 0/1.
// LDS x-tile is double-buffered: one barrier per K-step, staging load of
// tile k+1 overlaps the WMMA on tile k.
// ---------------------------------------------------------------------------
__global__ void proj_qk_kernel(const float* __restrict__ x,
                               const __bf16* __restrict__ wqb,
                               const float* __restrict__ bq,
                               const __bf16* __restrict__ wkb,
                               const float* __restrict__ bk,
                               __bf16* __restrict__ qb,
                               __bf16* __restrict__ kb) {
  int bidx = blockIdx.x;
  int b    = bidx >> 8;                      // 256 n-tiles per batch
  int n0   = (bidx & 255) * 16;
  int tid  = threadIdx.x, wave = tid >> 5, lane = tid & 31;

  const __bf16* W    = (wave < 2) ? wqb : wkb;
  const float*  bias = (wave < 2) ? bq : bk;
  __bf16*       outp = (wave < 2) ? qb : kb;
  int dtile = (wave & 1) * 16;

  __shared__ alignas(16) __bf16 xa[2][16][40];  // double-buffered, padded
  v8f acc = {};
  const float* xb_ = x + (size_t)b * Cch * Nn;
  int dcol = dtile + (lane & 15);

  store_xquad(load_xquad(xb_, 0, n0, tid), tid, xa[0]);
  __syncthreads();

#pragma unroll
  for (int it = 0; it < Cch / 32; ++it) {
    int k0  = it * 32;
    int buf = it & 1;

    // A = x^T tile (rows n, K = c): two contiguous 16B LDS runs per lane.
    int kg = (lane >> 4) * 8;
    const __bf16* arow = &xa[buf][lane & 15][kg];
    v16bf A = cat8(*(const v8bf*)arow, *(const v8bf*)(arow + 16));

    // B = W^T tile (K = c, cols d): contiguous bf16 run of W's row.
    const __bf16* wrow = W + (size_t)dcol * Cch + k0 + 8 * (lane >> 4);
    v16bf Bv = cat8(*(const v8bf*)wrow, *(const v8bf*)(wrow + 16));

    float4 nxt;
    if (it + 1 < Cch / 32) nxt = load_xquad(xb_, k0 + 32, n0, tid);

    acc = wmma_bf16(A, Bv, acc);

    if (it + 1 < Cch / 32) store_xquad(nxt, tid, xa[buf ^ 1]);
    __syncthreads();
  }

  float bia = bias[dcol];
#pragma unroll
  for (int v = 0; v < 8; ++v) {
    int nrow = n0 + v + 8 * (lane >> 4);
    outp[((size_t)b * Nn + nrow) * Dd + dcol] = (__bf16)(acc[v] + bia);
  }
}

// ---------------------------------------------------------------------------
// Kernel 4: v projection.  vb[b,c,n] = sum_cc Wv[c,cc]*x[b,cc,n] + bv[c]
// Weights pre-converted to bf16.  Stored bf16 [B,C,N].  Double-buffered LDS.
// ---------------------------------------------------------------------------
__global__ void proj_v_kernel(const float* __restrict__ x,
                              const __bf16* __restrict__ wvb,
                              const float* __restrict__ bv,
                              __bf16* __restrict__ vb) {
  int bidx = blockIdx.x;
  int b    = bidx >> 8;
  int n0   = (bidx & 255) * 16;
  int tid  = threadIdx.x, wave = tid >> 5, lane = tid & 31;

  __shared__ alignas(16) __bf16 xa[2][16][40];  // double-buffered, padded
  v8f acc[4] = {{}, {}, {}, {}};
  const float* xb_ = x + (size_t)b * Cch * Nn;

  store_xquad(load_xquad(xb_, 0, n0, tid), tid, xa[0]);
  __syncthreads();

#pragma unroll
  for (int it = 0; it < Cch / 32; ++it) {
    int k0  = it * 32;
    int buf = it & 1;

    // B = x tile (K = cc, cols n): lane col = lane&15, K runs at 8*(lane>>4) and +16
    const __bf16* brow = &xa[buf][lane & 15][8 * (lane >> 4)];
    v16bf Bv = cat8(*(const v8bf*)brow, *(const v8bf*)(brow + 16));

    float4 nxt;
    if (it + 1 < Cch / 32) nxt = load_xquad(xb_, k0 + 32, n0, tid);

#pragma unroll
    for (int t = 0; t < 4; ++t) {
      int c0   = (wave + 4 * t) * 16;
      int crow = c0 + (lane & 15);
      // A = Wv tile (rows c, K = cc): two contiguous 16B bf16 runs.
      const __bf16* wrow = wvb + (size_t)crow * Cch + k0 + (lane >> 4) * 8;
      v16bf A = cat8(*(const v8bf*)wrow, *(const v8bf*)(wrow + 16));
      acc[t] = wmma_bf16(A, Bv, acc[t]);
    }

    if (it + 1 < Cch / 32) store_xquad(nxt, tid, xa[buf ^ 1]);
    __syncthreads();
  }

#pragma unroll
  for (int t = 0; t < 4; ++t) {
    int c0   = (wave + 4 * t) * 16;
    int ncol = n0 + (lane & 15);
#pragma unroll
    for (int v = 0; v < 8; ++v) {
      int crow = c0 + v + 8 * (lane >> 4);
      vb[((size_t)b * Cch + crow) * Nn + ncol] = (__bf16)(acc[t][v] + bv[crow]);
    }
  }
}

// ---------------------------------------------------------------------------
// Kernel 5: fused attention + output combine.  M_TILE = 32 queries per block.
// 256 threads = 8 waves.  Softmax phases: wave w serves m-tile (w>>2) with
// n-tile stride 4.  GEMM phase: wave w owns c-tiles {w, w+8} and accumulates
// against BOTH m-tiles' p-panels, so each vb fragment is fetched once per
// block.  p panels: [32][4096] bf16 = 256KB of the 320KB WGP LDS.
// ---------------------------------------------------------------------------
__global__ void attn_kernel(const float* __restrict__ x,
                            const __bf16* __restrict__ qb,
                            const __bf16* __restrict__ kb,
                            const __bf16* __restrict__ vb,
                            const float* __restrict__ gamma,
                            const float* __restrict__ g,
                            float* __restrict__ out) {
  extern __shared__ __align__(16) __bf16 pbuf[];   // [32][4096] (256KB)
  __shared__ float redbuf[2][4][16];
  __shared__ float rcpsum[2][16];

  int bidx = blockIdx.x;
  int b    = bidx >> 7;                      // 128 m-blocks per batch
  int m0   = (bidx & 127) * 32;
  int tid  = threadIdx.x, wave = tid >> 5, lane = tid & 31;
  int grp  = wave >> 2;                      // which 16-row m-tile (0/1)
  int wv   = wave & 3;                       // role within group

  __bf16* ppanel = pbuf + grp * 16 * Nn;

  // q fragment A (rows m, K = d) for this group's m-tile.
  const __bf16* qrow = qb + ((size_t)b * Nn + m0 + grp * 16 + (lane & 15)) * Dd
                          + (lane >> 4) * 8;
  v16bf Aq = cat8(*(const v8bf*)qrow, *(const v8bf*)(qrow + 16));

  // ---- Phase 1: exact row max over all N ----
  float rmax[8];
#pragma unroll
  for (int v = 0; v < 8; ++v) rmax[v] = -3.0e38f;

  for (int nt = wv; nt < Nn / 16; nt += 4) {
    int n0 = nt * 16;
    const __bf16* krow = kb + ((size_t)b * Nn + n0 + (lane & 15)) * Dd + 8 * (lane >> 4);
    v16bf Bk = cat8(*(const v8bf*)krow, *(const v8bf*)(krow + 16));
    v8f S = {};
    S = wmma_bf16(Aq, Bk, S);
#pragma unroll
    for (int v = 0; v < 8; ++v) {
      float m_ = S[v];
      m_ = fmaxf(m_, __shfl_xor(m_, 1, 32));
      m_ = fmaxf(m_, __shfl_xor(m_, 2, 32));
      m_ = fmaxf(m_, __shfl_xor(m_, 4, 32));
      m_ = fmaxf(m_, __shfl_xor(m_, 8, 32));
      rmax[v] = fmaxf(rmax[v], m_);
    }
  }
  if ((lane & 15) == 0) {
#pragma unroll
    for (int v = 0; v < 8; ++v) redbuf[grp][wv][v + 8 * (lane >> 4)] = rmax[v];
  }
  __syncthreads();
#pragma unroll
  for (int v = 0; v < 8; ++v) {
    int row = v + 8 * (lane >> 4);
    rmax[v] = fmaxf(fmaxf(redbuf[grp][0][row], redbuf[grp][1][row]),
                    fmaxf(redbuf[grp][2][row], redbuf[grp][3][row]));
  }
  __syncthreads();                           // redbuf reused for sums

  // ---- Phase 2: p = exp(S - max) -> LDS, accumulate row sums ----
  float rsum[8] = {0.f, 0.f, 0.f, 0.f, 0.f, 0.f, 0.f, 0.f};
  for (int nt = wv; nt < Nn / 16; nt += 4) {
    int n0 = nt * 16;
    const __bf16* krow = kb + ((size_t)b * Nn + n0 + (lane & 15)) * Dd + 8 * (lane >> 4);
    v16bf Bk = cat8(*(const v8bf*)krow, *(const v8bf*)(krow + 16));
    v8f S = {};
    S = wmma_bf16(Aq, Bk, S);
#pragma unroll
    for (int v = 0; v < 8; ++v) {
      float p = expf(S[v] - rmax[v]);
      rsum[v] += p;
      int row = v + 8 * (lane >> 4);
      ppanel[row * Nn + n0 + (lane & 15)] = (__bf16)p;
    }
  }
#pragma unroll
  for (int v = 0; v < 8; ++v) {
    rsum[v] += __shfl_xor(rsum[v], 1, 32);
    rsum[v] += __shfl_xor(rsum[v], 2, 32);
    rsum[v] += __shfl_xor(rsum[v], 4, 32);
    rsum[v] += __shfl_xor(rsum[v], 8, 32);
  }
  if ((lane & 15) == 0) {
#pragma unroll
    for (int v = 0; v < 8; ++v) redbuf[grp][wv][v + 8 * (lane >> 4)] = rsum[v];
  }
  __syncthreads();
  if (tid < 32) {
    int gg = tid >> 4, row = tid & 15;
    float s_ = redbuf[gg][0][row] + redbuf[gg][1][row] +
               redbuf[gg][2][row] + redbuf[gg][3][row];
    rcpsum[gg][row] = 1.0f / s_;
  }
  __syncthreads();

  // ---- Phase 3: sp[c,m] = sum_n vb[c,n] * p[m,n], K = 4096 ----
  // wave owns c-tiles {wave, wave+8}; accumulates vs both m-tiles.
  v8f acc[2][2] = {{{}, {}}, {{}, {}}};
  const __bf16* vbase = vb + (size_t)b * Cch * Nn;
  for (int n0 = 0; n0 < Nn; n0 += 32) {
    v16bf Bp[2];
#pragma unroll
    for (int gg = 0; gg < 2; ++gg) {
      const __bf16* prow = pbuf + (gg * 16 + (lane & 15)) * Nn + n0 + 8 * (lane >> 4);
      Bp[gg] = cat8(*(const v8bf*)prow, *(const v8bf*)(prow + 16));
    }
    __builtin_prefetch(vbase + ((size_t)(wave * 16 + (lane & 15))) * Nn + n0 + 256, 0, 0);
#pragma unroll
    for (int t = 0; t < 2; ++t) {
      int c0 = (wave + 8 * t) * 16;
      const __bf16* vrow = vbase + ((size_t)(c0 + (lane & 15))) * Nn
                                 + n0 + (lane >> 4) * 8;
      v16bf Av = cat8(*(const v8bf*)vrow, *(const v8bf*)(vrow + 16));
      acc[t][0] = wmma_bf16(Av, Bp[0], acc[t][0]);
      acc[t][1] = wmma_bf16(Av, Bp[1], acc[t][1]);
    }
  }

  // ---- epilogue: out = gamma*sp/rowsum + (1 + lamb*a)*x ----
  float gam = gamma[0];
#pragma unroll
  for (int t = 0; t < 2; ++t) {
#pragma unroll
    for (int gg = 0; gg < 2; ++gg) {
      int   mcol = m0 + gg * 16 + (lane & 15);
      float rc   = rcpsum[gg][lane & 15];
#pragma unroll
      for (int v = 0; v < 8; ++v) {
        int crow = (wave + 8 * t) * 16 + v + 8 * (lane >> 4);
        size_t idx = ((size_t)b * Cch + crow) * Nn + mcol;
        out[idx] = gam * (acc[t][gg][v] * rc) + g[b * Cch + crow] * x[idx];
      }
    }
  }
}

// ---------------------------------------------------------------------------
extern "C" void kernel_launch(void* const* d_in, const int* in_sizes, int n_in,
                              void* d_out, int out_size, void* d_ws, size_t ws_size,
                              hipStream_t stream) {
  const float* x     = (const float*)d_in[0];
  const float* Wq    = (const float*)d_in[1];
  const float* bq    = (const float*)d_in[2];
  const float* Wk    = (const float*)d_in[3];
  const float* bk    = (const float*)d_in[4];
  const float* Wv    = (const float*)d_in[5];
  const float* bv    = (const float*)d_in[6];
  const float* gamma = (const float*)d_in[7];
  const float* W1    = (const float*)d_in[8];
  const float* W2    = (const float*)d_in[9];
  const float* lamb  = (const float*)d_in[10];
  float* out = (float*)d_out;

  // Workspace (~21.3 MB):
  //   qb(2MB) | kb(2MB) | vb(16MB) | wqb(16KB) | wkb(16KB) | wvb(128KB) | ...
  char* ws = (char*)d_ws;
  __bf16* qb  = (__bf16*)ws;
  __bf16* kb  = (__bf16*)(ws + (size_t)2 * 1024 * 1024);
  __bf16* vb  = (__bf16*)(ws + (size_t)4 * 1024 * 1024);
  __bf16* wqb = (__bf16*)(ws + (size_t)20 * 1024 * 1024);
  __bf16* wkb = wqb + Dd * Cch;
  __bf16* wvb = wkb + Dd * Cch;
  float*  meanv = (float*)(ws + (size_t)21 * 1024 * 1024);
  float*  stdv  = meanv + Bsz * Cch;
  float*  gbuf  = stdv  + Bsz * Cch;

  cvt_bf16_kernel<<<(Dd * Cch / 4 + 255) / 256, 256, 0, stream>>>(Wq, wqb, Dd * Cch);
  cvt_bf16_kernel<<<(Dd * Cch / 4 + 255) / 256, 256, 0, stream>>>(Wk, wkb, Dd * Cch);
  cvt_bf16_kernel<<<(Cch * Cch / 4 + 255) / 256, 256, 0, stream>>>(Wv, wvb, Cch * Cch);
  stats_kernel<<<Bsz * Cch, 256, 0, stream>>>(x, meanv, stdv);
  se_kernel<<<Bsz, 256, 0, stream>>>(meanv, stdv, W1, W2, lamb, gbuf);
  proj_qk_kernel<<<Bsz * (Nn / 16), 128, 0, stream>>>(x, wqb, bq, wkb, bk, qb, kb);
  proj_v_kernel<<<Bsz * (Nn / 16), 128, 0, stream>>>(x, wvb, bv, vb);
  attn_kernel<<<Bsz * (Nn / 32), 256, 32 * Nn * sizeof(__bf16), stream>>>(
      x, qb, kb, vb, gamma, gbuf, out);
}
